// VkWeightOnlyInt4Linear_47407849013686
// MI455X (gfx1250) — matrix operand order repaired and checked by reading, not compiled, not run on hardware
//
#include <hip/hip_runtime.h>

// Weight-only int4 GEMM for MI455X (gfx1250, wave32, WMMA).
// out[m,n] = sum_k X[m,k] * ((q[n,k]-8)*scale[g,n] + zero[g,n]),  G=128.
// M=2048, N=4096, K=4096 -> 68.7 GFLOP, ~96MB unique data (fits in 192MB L2).
// Compute-bound -> v_wmma_f32_16x16x32_f16 with 32x32 per-wave tiles so each
// A-build / B-ds_load feeds two WMMAs (amortizes f32->f16 cvt + LDS traffic).

typedef __attribute__((ext_vector_type(16))) _Float16 v16h;
typedef __attribute__((ext_vector_type(8)))  _Float16 v8h;
typedef __attribute__((ext_vector_type(8)))  float    v8f;

#define G_SIZE   128
#define BM       128  // block M tile (4 waves x 32)
#define BN       64   // block N tile (2 waves x 32)
#define BK       64
#define A_STRIDE 68   // floats/row: 64 + 4 pad -> 272B = 17*16 (b128-aligned rows)
#define B_STRIDE 80   // halves/row: 64 + 16 pad -> 160B = 10*16 (b128-aligned rows)

// gfx1250 async global->LDS DMA path (ASYNCcnt). Round-1 probe showed param 0
// is 'v4i addrspace(1)*' (typed).
#if __has_builtin(__builtin_amdgcn_global_load_async_to_lds_b128) && \
    __has_builtin(__builtin_amdgcn_s_wait_asynccnt)
#define USE_ASYNC_LDS 1
typedef int v4i __attribute__((vector_size(16)));
typedef __attribute__((address_space(1))) v4i gv4i;   // global 128-bit chunk
typedef __attribute__((address_space(3))) v4i lv4i;   // LDS 128-bit chunk
#endif

__global__ __launch_bounds__(256) void wo_int4_wmma_gemm(
    const float* __restrict__ X,   // [M][K]  activations (f32)
    const int*   __restrict__ Wp,  // [N][K/2] one packed byte per int32 (hi nibble = even k)
    const float* __restrict__ SZ,  // [K/G][N][2]  (scale, zero)
    float* __restrict__ Y,         // [M][N]
    int M, int N, int K)
{
    __shared__ float    As[BM][A_STRIDE];   // A tile kept f32 (async-DMA'd raw)
    __shared__ _Float16 Bs[BN][B_STRIDE];   // dequantized weights, f16

    const int tid   = threadIdx.x;
    const int lane  = tid & 31;
    const int wid   = tid >> 5;
    const int wm    = wid & 3;              // wave M sub-tile (0..3) -> 32 rows each
    const int wn    = wid >> 2;             // wave N sub-tile (0..1) -> 32 cols each
    const int lrow  = lane & 15;
    const bool hihf = (lane >= 16);

    const int mblk  = blockIdx.y * BM;
    const int nblk  = blockIdx.x * BN;
    const int khalf = K >> 1;

    v8f acc[2][2] = {};                     // [mt][nt] 16x16 f32 tiles

    for (int kb = 0; kb < K; kb += BK) {
        // ---- stage A tile: 128x64 f32 (32KB), 8 x b128 per thread ----
        const float* Ag = X + (size_t)mblk * K + kb;
#pragma unroll
        for (int i = 0; i < 8; ++i) {
            int idx = tid + i * 256;        // 0..2047 float4 slots
            int am  = idx >> 4;             // row 0..127
            int aq  = idx & 15;             // float4 within row
            const float* src = Ag + (size_t)am * K + aq * 4;
            float*       dst = &As[am][aq * 4];
#if defined(USE_ASYNC_LDS)
            __builtin_amdgcn_global_load_async_to_lds_b128(
                (gv4i*)(void*)src, (lv4i*)(void*)dst, 0, 0);
#else
            *(float4*)dst = *(const float4*)src;
#endif
        }

        // ---- stage B tile: dequant 64n x 32 packed-bytes -> 64n x 64k f16 ----
        // BK=64 never straddles a quant group (64 | 128): one group per chunk.
        const int    g  = kb / G_SIZE;
        const float* sz = SZ + ((size_t)g * N + nblk) * 2;
#pragma unroll
        for (int i = 0; i < 8; ++i) {
            int idx = tid + i * 256;        // 0..2047
            int bn  = idx >> 5;             // n within tile 0..63
            int kp  = idx & 31;             // k-pair within chunk 0..31
            int q   = Wp[(size_t)(nblk + bn) * khalf + (kb >> 1) + kp];
            float s = sz[(size_t)bn * 2 + 0];
            float z = sz[(size_t)bn * 2 + 1];
            float whi = (float)(((q >> 4) & 0xF) - 8) * s + z;  // even k
            float wlo = (float)((q        & 0xF) - 8) * s + z;  // odd  k
            union { _Float16 h[2]; unsigned u; } pk;
            pk.h[0] = (_Float16)whi;
            pk.h[1] = (_Float16)wlo;
            *(unsigned*)&Bs[bn][kp * 2] = pk.u;                 // 32-bit ds_store
        }

        // prefetch next chunk's packed weights into L2 (global_prefetch_b8)
        if (kb + BK < K)
            __builtin_prefetch(&Wp[(size_t)(nblk + (tid & 63)) * khalf + ((kb + BK) >> 1)], 0, 3);

#if defined(USE_ASYNC_LDS)
        __builtin_amdgcn_s_wait_asynccnt(0);
#endif
        __syncthreads();

        // ---- WMMA compute: 2 K-steps x 2 M x 2 N = 8 wmma per chunk ----
        // A 16x32 f16 as loaded here (contiguous k, same order as B): lanes 0-15 row=m, K c*32+0..15;
        //                     lanes 16-31 same rows, K c*32+16..31.
        const int akof = hihf ? 16 : 0;
        const int bkof = hihf ? 16 : 0;

        v16h a[2][2];                       // [mt][c]
#pragma unroll
        for (int mt = 0; mt < 2; ++mt) {
            const int arow = wm * 32 + mt * 16 + lrow;
#pragma unroll
            for (int c = 0; c < 2; ++c) {
                const float* ap = &As[arow][c * 32 + akof];
#pragma unroll
                for (int u = 0; u < 8; ++u) {
                    a[mt][c][u]     = (_Float16)ap[u];
                    a[mt][c][u + 8] = (_Float16)ap[u + 8];
                }
            }
        }

        // B 32x16 f16 as loaded here (contiguous k, same order as A): lane col = lane%16; lanes 0-15 K=0..15,
        // lanes 16-31 K=16..31 -> 16 contiguous halves = two b128 DS loads.
#pragma unroll
        for (int nt = 0; nt < 2; ++nt) {
            const int brow = wn * 32 + nt * 16 + lrow;
#pragma unroll
            for (int c = 0; c < 2; ++c) {
                const _Float16* bp = &Bs[brow][c * 32 + bkof];
                union { v16h v; v8h h[2]; } b;
                b.h[0] = *(const v8h*)(bp);
                b.h[1] = *(const v8h*)(bp + 8);
#pragma unroll
                for (int mt = 0; mt < 2; ++mt) {
                    acc[mt][nt] = __builtin_amdgcn_wmma_f32_16x16x32_f16(
                        false, a[mt][c], false, b.v, (short)0, acc[mt][nt],
                        false, false);
                }
            }
        }
        __syncthreads();
    }

    // ---- epilogue: C/D 16x16 f32 layout: vgpr r, lanes 0-15 -> M=r, N=lane;
    //      lanes 16-31 -> M=8+r, N=lane-16. ----
#pragma unroll
    for (int mt = 0; mt < 2; ++mt) {
        const int m0 = mblk + wm * 32 + mt * 16 + (hihf ? 8 : 0);
#pragma unroll
        for (int nt = 0; nt < 2; ++nt) {
            const int ng = nblk + wn * 32 + nt * 16 + lrow;
#pragma unroll
            for (int r = 0; r < 8; ++r) {
                Y[(size_t)(m0 + r) * N + ng] = acc[mt][nt][r];
            }
        }
    }
}

extern "C" void kernel_launch(void* const* d_in, const int* in_sizes, int n_in,
                              void* d_out, int out_size, void* d_ws, size_t ws_size,
                              hipStream_t stream) {
    const float* X  = (const float*)d_in[0];
    const int*   Wp = (const int*)d_in[1];
    const float* SZ = (const float*)d_in[2];
    float*       Y  = (float*)d_out;

    const int K = 4096;                                   // reference in_features
    const int N = (int)(((long long)in_sizes[1] * 2) / K);// 4096
    const int M = in_sizes[0] / K;                        // 2048 (batch*seq)

    dim3 grid(N / BN, M / BM);
    wo_int4_wmma_gemm<<<grid, 256, 0, stream>>>(X, Wp, SZ, Y, M, N, K);
}
